// CrossModalReinforcement_64115271795232
// MI455X (gfx1250) — compile-verified, matrix operand order checked by reading, not run on hardware
//
#include <hip/hip_runtime.h>
#include <hip/hip_bf16.h>
#include <math.h>

// ---------------------------------------------------------------------------
// CrossModalReinforcement for MI455X (gfx1250, wave32, WMMA 16x16x32 f16)
//
// - All matrix operands are pre-converted to f16 once (halves GEMM traffic).
// - GEMM: 64x64 tile / 128-thread workgroup; tiles staged into LDS by the
//   Tensor Data Mover (tensor_load_to_lds + s_wait_tensorcnt) when available,
//   manual f16 staging otherwise.  16 WMMAs per k-step per block.
// - Flash attention per 16-query tile, f16 operands, f32 softmax.
// - Sequential Mamba2 scan: 64x64 state in registers, LDS-staged operands.
// ---------------------------------------------------------------------------

typedef __attribute__((ext_vector_type(16))) _Float16 v16h;
typedef __attribute__((ext_vector_type(8)))  _Float16 v8h;
typedef __attribute__((ext_vector_type(8)))  float    v8f;
typedef __attribute__((ext_vector_type(4)))  unsigned int u32x4;
typedef __attribute__((ext_vector_type(8)))  int      i32x8;
typedef __attribute__((ext_vector_type(4)))  int      i32x4;

#define D_MODEL   512
#define L_SEQ     1024
#define BATCH     8
#define NTOK      (BATCH * L_SEQ)        // 8192
#define NUM_HEADS 8
#define HEAD_DIM  64
#define D_INNER   1024
#define NHEADS_M  16
#define HEADDIM_M 64
#define D_STATE   64
#define CONV_DIM  1152
#define D_IN_PROJ 2192
#define EPS_F     1e-5f

#if __has_builtin(__builtin_amdgcn_tensor_load_to_lds) && \
    __has_builtin(__builtin_amdgcn_s_wait_tensorcnt)
#define USE_TDM 1
#endif

__device__ __forceinline__ float silu_f(float x) {
    return x / (1.0f + __expf(-x));
}
__device__ __forceinline__ float softplus_f(float x) {
    return (x > 20.0f) ? x : log1pf(__expf(x));
}
__device__ __forceinline__ v16h cat8(v8h lo, v8h hi) {
    return __builtin_shufflevector(lo, hi, 0,1,2,3,4,5,6,7,8,9,10,11,12,13,14,15);
}

#ifdef USE_TDM
// Issue a 2D TDM tile load (f16 elements) into LDS.  Descriptor per
// CDNA5 ISA.  global addr points at tile start;
// tensor dims given as remaining extent from the tile start so HW OOB
// handling (reads beyond -> zero) covers the ragged last N-block.
// This toolchain exposes the 6-arg builtin form:
//   (u32x4 g0, i32x8 g1, i32x4 g2, i32x4 g3, i32x8 extra, i32 cpol)
__device__ __forceinline__ void tdm_load_2d_f16(unsigned lds_addr,
                                                const _Float16* gaddr,
                                                unsigned tensor_d0, unsigned tensor_d1,
                                                unsigned tile_d0, unsigned tile_d1,
                                                unsigned stride0_elems) {
    unsigned long long ga = (unsigned long long)(size_t)gaddr;
    u32x4 g0;
    g0[0] = 1u;                                        // count=1, user desc
    g0[1] = lds_addr;                                  // LDS byte address
    g0[2] = (unsigned)(ga & 0xFFFFFFFFu);              // global_addr[31:0]
    g0[3] = (unsigned)((ga >> 32) & 0x01FFFFFFu) | 0x80000000u;  // [56:32] | type=2
    i32x8 g1;
    g1[0] = (int)(1u << 16);                           // data_size = 1 -> 2 bytes
    g1[1] = (int)((tensor_d0 & 0xFFFFu) << 16);        // tensor_dim0[15:0]
    g1[2] = (int)((tensor_d0 >> 16) | ((tensor_d1 & 0xFFFFu) << 16));
    g1[3] = (int)((tensor_d1 >> 16) | (tile_d0 << 16));
    g1[4] = (int)(tile_d1 & 0xFFFFu);                  // tile_dim1, tile_dim2=0
    g1[5] = (int)stride0_elems;                        // tensor_dim0_stride lo
    g1[6] = 0;
    g1[7] = 0;
    i32x4 z4 = {0, 0, 0, 0};
    i32x8 z8 = {0, 0, 0, 0, 0, 0, 0, 0};
    __builtin_amdgcn_tensor_load_to_lds(g0, g1, z4, z4, z8, 0);
}
#endif

// ---------------------------------------------------------------------------
// f32 -> f16 conversion (vector of 4 per thread); n % 4 == 0.
// ---------------------------------------------------------------------------
__global__ void cvt_f32_f16_kernel(const float* __restrict__ src,
                                   _Float16* __restrict__ dst, int n) {
    int i = (blockIdx.x * blockDim.x + threadIdx.x) * 4;
    if (i >= n) return;
    float4 f = *(const float4*)(src + i);
    dst[i + 0] = (_Float16)f.x;
    dst[i + 1] = (_Float16)f.y;
    dst[i + 2] = (_Float16)f.z;
    dst[i + 3] = (_Float16)f.w;
}

// ---------------------------------------------------------------------------
// Tiled WMMA GEMM, f16 operands, f32 accumulation.
//   C[M,N] = A[M,K] * B[K,N] (+ residual);  output f32 or f16 (OUT16).
//   grid = (ceil(N/64), M/64), block = 128 (4 waves).
//   Requires M%64==0, K%32==0, N%16==0.
// LDS: lA = A chunk 64x32 row-major [m][k]  (contiguous fragment reads)
//      lB = B chunk 32x64 row-major [k][n]  (TDM-native layout; strided
//           fragment reads, conflict-free across lanes)
// ---------------------------------------------------------------------------
template <bool HAS_RES, bool OUT16>
__global__ void wmma_gemm_f16(const _Float16* __restrict__ A,
                              const _Float16* __restrict__ Bm,
                              float* __restrict__ Cf,
                              _Float16* __restrict__ Ch,
                              int M, int N, int K,
                              const float* __restrict__ residual) {
    __shared__ _Float16 lA[64 * 32];   // [m][k]
    __shared__ _Float16 lB[32 * 64];   // [k][n]

    const int tid  = threadIdx.x;      // 0..127
    const int w    = tid >> 5;         // wave 0..3
    const int lane = tid & 31;
    const int r    = lane & 15;
    const int hi   = lane >> 4;
    const int n0   = blockIdx.x * 64;
    const int m0   = blockIdx.y * 64;

#ifndef USE_TDM
    const int sa_m  = tid >> 1;          // 0..63
    const int sa_kb = (tid & 1) * 16;    // 0 or 16
    const int sb_k  = tid >> 2;          // 0..31
    const int sb_nb = (tid & 3) * 16;    // 0,16,32,48
#endif

    v8f acc[4] = {{}, {}, {}, {}};

    for (int k0 = 0; k0 < K; k0 += 32) {
#ifdef USE_TDM
        if (tid < 32) {   // wave 0 issues both DMA tiles, waits, others barrier
            tdm_load_2d_f16((unsigned)(size_t)&lA[0],
                            A + (size_t)m0 * K + k0,
                            (unsigned)(K - k0), (unsigned)(M - m0),
                            32u, 64u, (unsigned)K);
            tdm_load_2d_f16((unsigned)(size_t)&lB[0],
                            Bm + (size_t)k0 * N + n0,
                            (unsigned)(N - n0), (unsigned)(K - k0),
                            64u, 32u, (unsigned)N);
            __builtin_amdgcn_s_wait_tensorcnt(0);
        }
        __syncthreads();
#else
        {   // manual f16 staging
            const v8h* ap = (const v8h*)(A + (size_t)(m0 + sa_m) * K + k0 + sa_kb);
            *(v8h*)&lA[sa_m * 32 + sa_kb]     = ap[0];
            *(v8h*)&lA[sa_m * 32 + sa_kb + 8] = ap[1];
            v8h b0 = {}, b1 = {};
            if (n0 + sb_nb < N) {
                const v8h* bp = (const v8h*)(Bm + (size_t)(k0 + sb_k) * N + n0 + sb_nb);
                b0 = bp[0]; b1 = bp[1];
            }
            *(v8h*)&lB[sb_k * 64 + sb_nb]     = b0;
            *(v8h*)&lB[sb_k * 64 + sb_nb + 8] = b1;
        }
        __syncthreads();
#endif

        // A fragment (ISA 7.12.2): lane row r, halves0-7 k=hi*8+i, 8-15 k=16+hi*8+i
        const _Float16* ar = &lA[(w * 16 + r) * 32 + hi * 8];
        v16h afrag = cat8(*(const v8h*)ar, *(const v8h*)(ar + 16));
#pragma unroll
        for (int nt = 0; nt < 4; ++nt) {
            // B fragment: lane col r of n-tile nt; [k][n] layout -> stride 64
            const _Float16* bp = &lB[(hi * 8) * 64 + nt * 16 + r];
            v16h bfrag;
#pragma unroll
            for (int i = 0; i < 8; ++i) {
                bfrag[i]     = bp[i * 64];
                bfrag[i + 8] = bp[(16 + i) * 64];
            }
            acc[nt] = __builtin_amdgcn_wmma_f32_16x16x32_f16(
                false, afrag, false, bfrag, (short)0, acc[nt], false, false);
        }
        __syncthreads();
    }

    // store: D layout lane -> N=r, M=hi*8+v
#pragma unroll
    for (int nt = 0; nt < 4; ++nt) {
        int n = n0 + nt * 16 + r;
        if (n >= N) continue;
#pragma unroll
        for (int v = 0; v < 8; ++v) {
            size_t idx = (size_t)(m0 + w * 16 + hi * 8 + v) * N + n;
            float val = acc[nt][v];
            if (HAS_RES) val += residual[idx];
            if (OUT16) Ch[idx] = (_Float16)val;
            else       Cf[idx] = val;
        }
    }
}

// ---------------------------------------------------------------------------
// Flash attention, one wave per (16-query tile, head, batch), f16 Q/K/V.
// grid = (L/16, NUM_HEADS, BATCH), block = 32.  Output ctx f16.
// ---------------------------------------------------------------------------
__global__ void attn_kernel(const _Float16* __restrict__ Q,
                            const _Float16* __restrict__ K,
                            const _Float16* __restrict__ V,
                            _Float16* __restrict__ ctx) {
    const int lane = threadIdx.x;
    const int r    = lane & 15;
    const int hi   = lane >> 4;
    const int qt   = blockIdx.x;
    const int h    = blockIdx.y;
    const int bb   = blockIdx.z;
    const size_t base = (size_t)bb * L_SEQ * D_MODEL + (size_t)h * HEAD_DIM;
    const float scale = 0.125f;  // 1/sqrt(64), applied to f32 scores

    v16h qa[2];
#pragma unroll
    for (int s = 0; s < 2; ++s) {
        const _Float16* qrow = Q + base + (size_t)(qt * 16 + r) * D_MODEL + s * 32 + hi * 8;
        qa[s] = cat8(*(const v8h*)qrow, *(const v8h*)(qrow + 16));
    }

    float m_i[8], l_i[8];
    v8f o_acc[4] = {{}, {}, {}, {}};
#pragma unroll
    for (int v = 0; v < 8; ++v) { m_i[v] = -1e30f; l_i[v] = 0.0f; }

    __shared__ float s_p[16 * 32];

    for (int kt = 0; kt < L_SEQ; kt += 32) {
        v8f s0 = {}, s1 = {};
#pragma unroll
        for (int s = 0; s < 2; ++s) {
            const _Float16* k0p = K + base + (size_t)(kt + r) * D_MODEL + s * 32 + hi * 8;
            const _Float16* k1p = K + base + (size_t)(kt + 16 + r) * D_MODEL + s * 32 + hi * 8;
            v16h b0 = cat8(*(const v8h*)k0p, *(const v8h*)(k0p + 16));
            v16h b1 = cat8(*(const v8h*)k1p, *(const v8h*)(k1p + 16));
            s0 = __builtin_amdgcn_wmma_f32_16x16x32_f16(false, qa[s], false, b0, (short)0, s0, false, false);
            s1 = __builtin_amdgcn_wmma_f32_16x16x32_f16(false, qa[s], false, b1, (short)0, s1, false, false);
        }

        float p0[8], p1[8], corr[8];
#pragma unroll
        for (int v = 0; v < 8; ++v) {
            float sv0 = s0[v] * scale;
            float sv1 = s1[v] * scale;
            float mx = fmaxf(sv0, sv1);
#pragma unroll
            for (int x = 1; x < 16; x <<= 1) mx = fmaxf(mx, __shfl_xor(mx, x, 32));
            float mnew = fmaxf(m_i[v], mx);
            corr[v] = __expf(m_i[v] - mnew);
            float e0 = __expf(sv0 - mnew);
            float e1 = __expf(sv1 - mnew);
            float rs = e0 + e1;
#pragma unroll
            for (int x = 1; x < 16; x <<= 1) rs += __shfl_xor(rs, x, 32);
            l_i[v] = l_i[v] * corr[v] + rs;
            m_i[v] = mnew;
            p0[v] = e0;
            p1[v] = e1;
        }
#pragma unroll
        for (int t = 0; t < 4; ++t)
#pragma unroll
            for (int v = 0; v < 8; ++v) o_acc[t][v] *= corr[v];

#pragma unroll
        for (int v = 0; v < 8; ++v) {
            s_p[(hi * 8 + v) * 32 + r]      = p0[v];
            s_p[(hi * 8 + v) * 32 + 16 + r] = p1[v];
        }
        __syncthreads();
        v16h pa;
#pragma unroll
        for (int i = 0; i < 8; ++i) {
            pa[i]     = (_Float16)s_p[r * 32 + hi * 8 + i];
            pa[i + 8] = (_Float16)s_p[r * 32 + 16 + hi * 8 + i];
        }
        __syncthreads();

#pragma unroll
        for (int t = 0; t < 4; ++t) {
            v16h vb;
            const _Float16* vtop = V + base + (size_t)(kt + hi * 8) * D_MODEL + t * 16 + r;
#pragma unroll
            for (int i = 0; i < 8; ++i) {
                vb[i]     = vtop[(size_t)i * D_MODEL];
                vb[i + 8] = vtop[(size_t)(16 + i) * D_MODEL];
            }
            o_acc[t] = __builtin_amdgcn_wmma_f32_16x16x32_f16(
                false, pa, false, vb, (short)0, o_acc[t], false, false);
        }
    }

#pragma unroll
    for (int t = 0; t < 4; ++t) {
#pragma unroll
        for (int v = 0; v < 8; ++v) {
            float inv = 1.0f / l_i[v];
            size_t row = (size_t)qt * 16 + hi * 8 + v;
            ctx[base + row * D_MODEL + t * 16 + r] = (_Float16)(o_acc[t][v] * inv);
        }
    }
}

// ---------------------------------------------------------------------------
// RMSNorm over 512, f32 in -> f16 out.  grid = NTOK, block = 128.
// ---------------------------------------------------------------------------
__global__ void rmsnorm512_kernel(const float* __restrict__ x,
                                  const float* __restrict__ w,
                                  _Float16* __restrict__ out) {
    const int row = blockIdx.x;
    const int t   = threadIdx.x;
    const float* xr = x + (size_t)row * D_MODEL;
    float v[4];
    float ss = 0.0f;
#pragma unroll
    for (int i = 0; i < 4; ++i) {
        v[i] = xr[t + i * 128];
        ss += v[i] * v[i];
    }
    __shared__ float red[128];
    red[t] = ss;
    __syncthreads();
    for (int s = 64; s > 0; s >>= 1) {
        if (t < s) red[t] += red[t + s];
        __syncthreads();
    }
    float inv = rsqrtf(red[0] / (float)D_MODEL + EPS_F);
    _Float16* orow = out + (size_t)row * D_MODEL;
#pragma unroll
    for (int i = 0; i < 4; ++i)
        orow[t + i * 128] = (_Float16)(v[i] * inv * w[t + i * 128]);
}

// ---------------------------------------------------------------------------
// Causal depthwise conv1d (taps=4) + bias + SiLU over CONV_DIM channels.
// ---------------------------------------------------------------------------
__global__ void conv_silu_kernel(const float* __restrict__ zx,
                                 const float* __restrict__ cw,
                                 const float* __restrict__ cb,
                                 float* __restrict__ xbc) {
    int idx = blockIdx.x * blockDim.x + threadIdx.x;
    if (idx >= BATCH * CONV_DIM) return;
    const int bb = idx / CONV_DIM;
    const int c  = idx % CONV_DIM;
    const float w0 = cw[c * 4 + 0], w1 = cw[c * 4 + 1];
    const float w2 = cw[c * 4 + 2], w3 = cw[c * 4 + 3];
    const float bias = cb[c];
    float x0 = 0.f, x1 = 0.f, x2 = 0.f;
    const float* src = zx + (size_t)bb * L_SEQ * D_IN_PROJ + D_INNER + c;
    float* dst = xbc + (size_t)bb * L_SEQ * CONV_DIM + c;
    for (int t = 0; t < L_SEQ; ++t) {
        float x3 = src[(size_t)t * D_IN_PROJ];
        float y = bias + w0 * x0 + w1 * x1 + w2 * x2 + w3 * x3;
        dst[(size_t)t * CONV_DIM] = silu_f(y);
        x0 = x1; x1 = x2; x2 = x3;
    }
}

// ---------------------------------------------------------------------------
// dt = softplus(dt_raw + dt_bias); dt_raw in zxbcdt cols [2176,2192).
// ---------------------------------------------------------------------------
__global__ void dt_softplus_kernel(const float* __restrict__ zx,
                                   const float* __restrict__ dt_bias,
                                   float* __restrict__ dt_out) {
    int idx = blockIdx.x * blockDim.x + threadIdx.x;
    if (idx >= NTOK * NHEADS_M) return;
    int row = idx / NHEADS_M;
    int hh  = idx % NHEADS_M;
    float x = zx[(size_t)row * D_IN_PROJ + (D_INNER + CONV_DIM) + hh] + dt_bias[hh];
    dt_out[idx] = softplus_f(x);
}

// ---------------------------------------------------------------------------
// Sequential SSM scan.  grid = (NHEADS_M, BATCH), block = 256.
// ---------------------------------------------------------------------------
__global__ void ssm_scan_kernel(const float* __restrict__ xbc,
                                const float* __restrict__ dt,
                                const float* __restrict__ A_log,
                                const float* __restrict__ Dp,
                                float* __restrict__ y) {
    const int h  = blockIdx.x;
    const int bb = blockIdx.y;
    const int t  = threadIdx.x;
    const int p  = t & 63;
    const int ng = t >> 6;
    const float A  = -__expf(A_log[h]);
    const float Dh = Dp[h];

    float hs[16];
#pragma unroll
    for (int j = 0; j < 16; ++j) hs[j] = 0.0f;

    __shared__ float sx[64], sB[64], sC[64], spart[4 * 64];

    for (int step = 0; step < L_SEQ; ++step) {
        const size_t row = (size_t)bb * L_SEQ + step;
        const float* xr = xbc + row * CONV_DIM;
        if (t < 64)        sx[t]        = xr[h * HEADDIM_M + t];
        else if (t < 128)  sB[t - 64]   = xr[D_INNER + (t - 64)];
        else if (t < 192)  sC[t - 128]  = xr[D_INNER + D_STATE + (t - 128)];
        __syncthreads();

        const float dtv = dt[row * NHEADS_M + h];
        const float dA  = __expf(dtv * A);
        const float xp  = sx[p];
        const float dBx = dtv * xp;
        float part = 0.0f;
#pragma unroll
        for (int j = 0; j < 16; ++j) {
            int n = ng * 16 + j;
            hs[j] = hs[j] * dA + dBx * sB[n];
            part += hs[j] * sC[n];
        }
        spart[ng * 64 + p] = part;
        __syncthreads();
        if (ng == 0) {
            float yv = spart[p] + spart[64 + p] + spart[128 + p] + spart[192 + p]
                     + Dh * xp;
            y[row * D_INNER + h * HEADDIM_M + p] = yv;
        }
        __syncthreads();
    }
}

// ---------------------------------------------------------------------------
// Gated RMSNorm over D_INNER = 1024, f16 out: g = y*silu(z); rmsnorm(g)*w.
// ---------------------------------------------------------------------------
__global__ void gate_norm_kernel(const float* __restrict__ y,
                                 const float* __restrict__ zx,
                                 const float* __restrict__ w,
                                 _Float16* __restrict__ out) {
    const int row = blockIdx.x;
    const int t   = threadIdx.x;
    const float* yr = y + (size_t)row * D_INNER;
    const float* zr = zx + (size_t)row * D_IN_PROJ;
    float g[4];
    float ss = 0.0f;
#pragma unroll
    for (int i = 0; i < 4; ++i) {
        int c = t + i * 256;
        g[i] = yr[c] * silu_f(zr[c]);
        ss += g[i] * g[i];
    }
    __shared__ float red[256];
    red[t] = ss;
    __syncthreads();
    for (int s = 128; s > 0; s >>= 1) {
        if (t < s) red[t] += red[t + s];
        __syncthreads();
    }
    float inv = rsqrtf(red[0] / (float)D_INNER + EPS_F);
    _Float16* orow = out + (size_t)row * D_INNER;
#pragma unroll
    for (int i = 0; i < 4; ++i) {
        int c = t + i * 256;
        orow[c] = (_Float16)(g[i] * inv * w[c]);
    }
}

// ---------------------------------------------------------------------------
// Host-side launcher
// ---------------------------------------------------------------------------
extern "C" void kernel_launch(void* const* d_in, const int* in_sizes, int n_in,
                              void* d_out, int out_size, void* d_ws, size_t ws_size,
                              hipStream_t stream) {
    const float* x_source = (const float*)d_in[0];
    const float* x_target = (const float*)d_in[1];
    const float* Wq       = (const float*)d_in[2];
    const float* Wk       = (const float*)d_in[3];
    const float* Wv       = (const float*)d_in[4];
    const float* Wo       = (const float*)d_in[5];
    const float* rms_w    = (const float*)d_in[6];
    const float* W_in     = (const float*)d_in[7];
    const float* conv_w   = (const float*)d_in[8];
    const float* conv_b   = (const float*)d_in[9];
    const float* dt_bias  = (const float*)d_in[10];
    const float* A_log    = (const float*)d_in[11];
    const float* D_param  = (const float*)d_in[12];
    const float* gnorm_w  = (const float*)d_in[13];
    const float* W_out    = (const float*)d_in[14];
    float* out = (float*)d_out;

    // ---- workspace layout (f16 region first, then f32 region; ~241 MB) ----
    _Float16* h = (_Float16*)d_ws;
    const size_t T512 = (size_t)NTOK * D_MODEL;
    _Float16* xs16   = h;               h += T512;              // x_source f16
    _Float16* xt16   = h;               h += T512;              // x_target f16
    _Float16* q16    = h;               h += T512;
    _Float16* k16    = h;               h += T512;
    _Float16* v16    = h;               h += T512;
    _Float16* ctx16  = h;               h += T512;
    _Float16* min16  = h;               h += T512;              // mamba_in f16
    _Float16* norm16 = h;               h += (size_t)NTOK * D_INNER;
    _Float16* wq16   = h;               h += (size_t)D_MODEL * D_MODEL;
    _Float16* wk16   = h;               h += (size_t)D_MODEL * D_MODEL;
    _Float16* wv16   = h;               h += (size_t)D_MODEL * D_MODEL;
    _Float16* wo16   = h;               h += (size_t)D_MODEL * D_MODEL;
    _Float16* win16  = h;               h += (size_t)D_MODEL * D_IN_PROJ;
    _Float16* wout16 = h;               h += (size_t)D_INNER * D_MODEL;
    float* f = (float*)h;
    float* attn_buf = f;                f += T512;
    float* zx_buf   = f;                f += (size_t)NTOK * D_IN_PROJ;
    float* xbc_buf  = f;                f += (size_t)NTOK * CONV_DIM;
    float* dt_buf   = f;                f += (size_t)NTOK * NHEADS_M;
    float* y_buf    = f;                f += (size_t)NTOK * D_INNER;

    dim3 blk128(128);
    dim3 g512(D_MODEL / 64, NTOK / 64);                 // (8, 128)
    dim3 gin((D_IN_PROJ + 63) / 64, NTOK / 64);         // (35, 128)

    // ---- 0. one-time f16 conversions (activations inputs + all weights) ----
    auto cvt = [&](const float* s, _Float16* d, size_t n) {
        cvt_f32_f16_kernel<<<(unsigned)((n / 4 + 255) / 256), 256, 0, stream>>>(s, d, (int)n);
    };
    cvt(x_source, xs16, T512);
    cvt(x_target, xt16, T512);
    cvt(Wq, wq16, (size_t)D_MODEL * D_MODEL);
    cvt(Wk, wk16, (size_t)D_MODEL * D_MODEL);
    cvt(Wv, wv16, (size_t)D_MODEL * D_MODEL);
    cvt(Wo, wo16, (size_t)D_MODEL * D_MODEL);
    cvt(W_in, win16, (size_t)D_MODEL * D_IN_PROJ);
    cvt(W_out, wout16, (size_t)D_INNER * D_MODEL);

    // ---- 1-3. Q/K/V projections (f16 out, consumed by attention) ----
    wmma_gemm_f16<false, true><<<g512, blk128, 0, stream>>>(
        xs16, wq16, nullptr, q16, NTOK, D_MODEL, D_MODEL, nullptr);
    wmma_gemm_f16<false, true><<<g512, blk128, 0, stream>>>(
        xt16, wk16, nullptr, k16, NTOK, D_MODEL, D_MODEL, nullptr);
    wmma_gemm_f16<false, true><<<g512, blk128, 0, stream>>>(
        xt16, wv16, nullptr, v16, NTOK, D_MODEL, D_MODEL, nullptr);

    // ---- 4. flash attention -> ctx (f16) ----
    attn_kernel<<<dim3(L_SEQ / 16, NUM_HEADS, BATCH), dim3(32), 0, stream>>>(
        q16, k16, v16, ctx16);

    // ---- 5. attn_out = ctx @ Wo (f32) ----
    wmma_gemm_f16<false, false><<<g512, blk128, 0, stream>>>(
        ctx16, wo16, attn_buf, nullptr, NTOK, D_MODEL, D_MODEL, nullptr);

    // ---- 6. mamba_in = rmsnorm(attn_out) (f16) ----
    rmsnorm512_kernel<<<NTOK, 128, 0, stream>>>(attn_buf, rms_w, min16);

    // ---- 7. zxbcdt = mamba_in @ W_in (f32, [8192 x 2192]) ----
    wmma_gemm_f16<false, false><<<gin, blk128, 0, stream>>>(
        min16, win16, zx_buf, nullptr, NTOK, D_IN_PROJ, D_MODEL, nullptr);

    // ---- 8. causal depthwise conv + silu ----
    conv_silu_kernel<<<(BATCH * CONV_DIM + 255) / 256, 256, 0, stream>>>(
        zx_buf, conv_w, conv_b, xbc_buf);

    // ---- 9. dt = softplus(dt + dt_bias) ----
    dt_softplus_kernel<<<(NTOK * NHEADS_M + 255) / 256, 256, 0, stream>>>(
        zx_buf, dt_bias, dt_buf);

    // ---- 10. sequential SSM scan -> y ----
    ssm_scan_kernel<<<dim3(NHEADS_M, BATCH), 256, 0, stream>>>(
        xbc_buf, dt_buf, A_log, D_param, y_buf);

    // ---- 11. gated RMSNorm (f16 out) ----
    gate_norm_kernel<<<NTOK, 256, 0, stream>>>(y_buf, zx_buf, gnorm_w, norm16);

    // ---- 12. out = norm @ W_out + x_source (residual fused, f32) ----
    wmma_gemm_f16<true, false><<<g512, blk128, 0, stream>>>(
        norm16, wout16, out, nullptr, NTOK, D_MODEL, D_INNER, x_source);
}
